// WD_GCN_reg_21878563406450
// MI455X (gfx1250) — 8-bit matrix operand order repaired and checked by reading, compiled, not run on hardware
//
#include <hip/hip_runtime.h>


#define T_DIM 32
#define N_DIM 20000
#define F_DIM 128
#define H_DIM 128

typedef __attribute__((ext_vector_type(16))) int   v16i;
typedef __attribute__((ext_vector_type(8)))  float v8f;

constexpr unsigned OFF_SB  = 0;
constexpr unsigned OFF_SW  = 131072;
constexpr unsigned OFF_SY  = 147456;
constexpr unsigned OFF_SH  = 155648;
constexpr unsigned OFF_OUT = 163840;
constexpr unsigned SMEM_TOTAL = 164352;

constexpr float BSCALE = 16.0f;
constexpr float ASCALE = 4.0f;
constexpr float DESC   = 1.0f / 64.0f;
constexpr float YSTORE = ASCALE / BSCALE;
constexpr float LOG2E  = 1.44269504088896340736f;
constexpr float NDL    = -DESC * LOG2E;
constexpr float HDESC  = 0.5f * DESC;

#if __has_builtin(__builtin_amdgcn_cvt_pk_fp8_f32)
#define HW_FP8 1
__device__ __forceinline__ unsigned char f2fp8(float v) {
  return (unsigned char)(__builtin_amdgcn_cvt_pk_fp8_f32(v, v, 0, false) & 0xFF);
}
#else
#define HW_FP8 0
__device__ __forceinline__ unsigned char f2fp8(float x) {
  float ax = fabsf(x);
  ax = fminf(ax, 448.0f);
  unsigned u = __builtin_bit_cast(unsigned, ax);
  unsigned r = u + 0x0007FFFFu + ((u >> 20) & 1u);
  int e = (int)(r >> 23) - 120;
  unsigned m = (r >> 20) & 7u;
  unsigned mag = (e <= 0) ? 0u : (((unsigned)e << 3) | m);
  if (e > 15) mag = 0x7Eu;
  return (unsigned char)(mag | ((x < 0.0f) ? 0x80u : 0u));
}
#endif

#if __has_builtin(__builtin_amdgcn_rcpf)
__device__ __forceinline__ float frcp(float x) { return __builtin_amdgcn_rcpf(x); }
#else
__device__ __forceinline__ float frcp(float x) { return 1.0f / x; }
#endif
#if __has_builtin(__builtin_amdgcn_exp2f)
__device__ __forceinline__ float fexp2(float x) { return __builtin_amdgcn_exp2f(x); }
#else
__device__ __forceinline__ float fexp2(float x) { return exp2f(x); }
#endif
#if __has_builtin(__builtin_amdgcn_fmed3f)
__device__ __forceinline__ float frelu(float x) { return __builtin_amdgcn_fmed3f(x, 0.0f, 3.0e38f); }
#else
__device__ __forceinline__ float frelu(float x) { return x > 0.0f ? x : 0.0f; }
#endif

#if __has_builtin(__builtin_amdgcn_tanhf)
#define HAS_TANH 1
__device__ __forceinline__ float ftanh(float x) { return __builtin_amdgcn_tanhf(x); }
#else
#define HAS_TANH 0
#endif

__device__ __forceinline__ float fsig2(float u) { return frcp(1.0f + fexp2(u)); }

#define WMMA_FP8(A, B, C) \
  __builtin_amdgcn_wmma_f32_16x16x128_fp8_fp8((A), (B), (short)0, (C), false, false)


__global__ __launch_bounds__(256, 1) void wdgcn_fused_fp8_kernel(
    const float* __restrict__ AX, const float* __restrict__ W,
    const float* __restrict__ Wf, const float* __restrict__ Wj,
    const float* __restrict__ Wc, const float* __restrict__ Wo,
    const float* __restrict__ Uf, const float* __restrict__ Uj,
    const float* __restrict__ Uc, const float* __restrict__ Uo,
    const float* __restrict__ bfv, const float* __restrict__ bjv,
    const float* __restrict__ bcv, const float* __restrict__ bov,
    const float* __restrict__ h_init, const float* __restrict__ c_init,
    const float* __restrict__ lin_w, const float* __restrict__ lin_b,
    float* __restrict__ out) {
  extern __shared__ char smem[];

  const int tid    = threadIdx.x;
  const int lane   = tid & 31;
  const int wvid   = tid >> 5;
  const int tile   = wvid >> 1;
  const int half   = wvid & 1;
  const int lane16 = lane & 15;
  const int hi     = lane >> 4;
  const int htb    = half * 4;

  for (int idx = tid; idx < 256 * 512; idx += 256) {
    int k = idx >> 9, g = idx & 511;
    int gate = g >> 7, col = g & 127;
    int kpart = k >> 7, krow = k & 127;
    const float* src;
    if (kpart == 0)
      src = (gate == 0 ? Wf : gate == 1 ? Wj : gate == 2 ? Wo : Wc) + krow * 128 + col;
    else
      src = (gate == 0 ? Uf : gate == 1 ? Uj : gate == 2 ? Uo : Uc) + krow * 128 + col;
    unsigned char v = f2fp8(*src * BSCALE);
    int ht = col >> 4, gcol = col & 15;
    int fragid = (kpart * 4 + gate) * 8 + ht;
    int kin = krow & 31;
    int dl = gcol + ((krow & 8) << 1);
    int byteo = (krow >> 6) * 32 + ((krow >> 4) & 3) * 8 + (krow & 7);
    *(unsigned char*)(smem + OFF_SB + (unsigned)(fragid * 2048 + dl * 64 + byteo)) = v;
  }
  for (int idx = tid; idx < 128 * 128; idx += 256) {
    int k = idx >> 7, c = idx & 127;
    unsigned char v = f2fp8(W[k * 128 + c] * BSCALE);
    int ct = c >> 4, gcol = c & 15;
    int kin = k & 31;
    int dl = gcol + ((k & 8) << 1);
    int byteo = (k >> 6) * 32 + ((k >> 4) & 3) * 8 + (k & 7);
    *(unsigned char*)(smem + OFF_SW + (unsigned)(ct * 2048 + dl * 64 + byteo)) = v;
  }
  for (int idx = tid; idx < 4 * 2048; idx += 256) {
    int b = idx & 2047;
    int dl = b >> 6, byteo = b & 63;
    int b2 = byteo & 31;
    int K = (byteo >> 5) * 64 + (b2 >> 3) * 16 + ((dl & 16) >> 1) + (b2 & 7);
    *(unsigned char*)(smem + OFF_SH + (unsigned)idx) = f2fp8(h_init[K] * ASCALE);
  }

  float linreg[4], breg[4][4];
  v8f cacc[4];
#pragma unroll
  for (int i = 0; i < 4; ++i) {
    int colx = (htb + i) * 16 + lane16;
    linreg[i] = lin_w[colx] * (1.0f / ASCALE);
#if HAS_TANH
    breg[0][i] = bfv[colx] * 0.5f;
    breg[1][i] = bjv[colx] * 0.5f;
    breg[2][i] = bov[colx] * 0.5f;
    breg[3][i] = bcv[colx] * 0.5f;
#else
    breg[0][i] = bfv[colx] * (-LOG2E);
    breg[1][i] = bjv[colx] * (-LOG2E);
    breg[2][i] = bov[colx] * (-LOG2E) - 2.0f;
    breg[3][i] = bcv[colx] * (-LOG2E);
#endif
    float cv = c_init[colx];
#pragma unroll
    for (int r = 0; r < 8; ++r) cacc[i][r] = cv;
  }
  const float lb = lin_b[0];

  __syncthreads();

  const int nrow0  = blockIdx.x * 64 + tile * 16;
  const int myrow  = nrow0 + lane16;
  const int nclamp = myrow < N_DIM ? myrow : (N_DIM - 1);
  char* sy = smem + OFF_SY + tile * 2048;
  char* sh = smem + OFF_SH + tile * 2048;

#pragma unroll 1
  for (int t = 0; t < T_DIM; ++t) {
    const float* axp = AX + ((size_t)t * N_DIM + (size_t)nclamp) * F_DIM;
    v16i axf;
#pragma unroll
    for (int d = 0; d < 16; ++d) {
      int K0 = (d >> 3) * 64 + ((d >> 1) & 3) * 16 + hi * 8 + (d & 1) * 4;
      float4 x = *(const float4*)(axp + K0);
#if HW_FP8
      int p = __builtin_amdgcn_cvt_pk_fp8_f32(x.x, x.y, 0, false);
      p     = __builtin_amdgcn_cvt_pk_fp8_f32(x.z, x.w, p, true);
#else
      int p = (int)f2fp8(x.x) | ((int)f2fp8(x.y) << 8) |
              ((int)f2fp8(x.z) << 16) | ((int)f2fp8(x.w) << 24);
#endif
      axf[d] = p;
    }
#pragma unroll
    for (int ci = 0; ci < 4; ++ci) {
      int ct = htb + ci;
      v16i bw = *(const v16i*)(smem + OFF_SW + (unsigned)(ct * 2048 + lane * 64));
      v8f acc = {};
      acc = WMMA_FP8(axf, bw, acc);
      int K = ct * 16 + lane16;
      int byteo = (K >> 6) * 32 + (((K & 63) >> 4) << 3) + (K & 7);
      char* base = sy + byteo + ((K & 8) ? 16 * 64 : 0);
#pragma unroll
      for (int r = 0; r < 8; ++r) {
        float v = frelu(acc[r] * YSTORE);
        *(unsigned char*)(base + (r + hi * 8) * 64) = f2fp8(v);
      }
    }
    __builtin_amdgcn_sched_barrier(0);

    __syncthreads();

    v16i yA = *(const v16i*)(sy + lane * 64);
    v16i hA = *(const v16i*)(sh + lane * 64);
    if (t > 0 && half == 0 && lane < 16) {
      float2 p = *(const float2*)(smem + OFF_OUT + (unsigned)((tile * 16 + lane16) * 8));
      int n = nrow0 + lane16;
      if (n < N_DIM) out[(size_t)(t - 1) * N_DIM + n] = p.x + p.y + lb;
    }

    __syncthreads();

    v8f outacc = {};
#pragma unroll
    for (int i = 0; i < 4; ++i) {
      int ht = htb + i;
      const char* bb = smem + OFF_SB + (unsigned)(ht * 2048 + lane * 64);
      v8f g0 = {}, g1 = {}, g2 = {}, g3 = {};
      {
        v16i b0 = *(const v16i*)(bb + 0 * 16384);
        v16i b1 = *(const v16i*)(bb + 1 * 16384);
        v16i b2 = *(const v16i*)(bb + 2 * 16384);
        v16i b3 = *(const v16i*)(bb + 3 * 16384);
        g0 = WMMA_FP8(yA, b0, g0);
        g1 = WMMA_FP8(yA, b1, g1);
        g2 = WMMA_FP8(yA, b2, g2);
        g3 = WMMA_FP8(yA, b3, g3);
      }
      {
        v16i b0 = *(const v16i*)(bb + 65536 + 0 * 16384);
        v16i b1 = *(const v16i*)(bb + 65536 + 1 * 16384);
        v16i b2 = *(const v16i*)(bb + 65536 + 2 * 16384);
        v16i b3 = *(const v16i*)(bb + 65536 + 3 * 16384);
        g0 = WMMA_FP8(hA, b0, g0);
        g1 = WMMA_FP8(hA, b1, g1);
        g2 = WMMA_FP8(hA, b2, g2);
        g3 = WMMA_FP8(hA, b3, g3);
      }
      int K = ht * 16 + lane16;
      int byteo = (K >> 6) * 32 + (((K & 63) >> 4) << 3) + (K & 7);
      char* baseh = sh + byteo + ((K & 8) ? 16 * 64 : 0);
#pragma unroll
      for (int r = 0; r < 8; ++r) {
#if HAS_TANH
        float tf = ftanh(fmaf(g0[r], HDESC, breg[0][i]));
        float tj = ftanh(fmaf(g1[r], HDESC, breg[1][i]));
        float to = ftanh(fmaf(g2[r], HDESC, breg[2][i]));
        float tc = ftanh(fmaf(g3[r], HDESC, breg[3][i]));
        float fg  = fmaf(tf, 0.5f, 0.5f);
        float jg  = fmaf(tj, 0.5f, 0.5f);
        float og4 = fmaf(to, 2.0f, 2.0f);
        float cg  = fmaf(tc, 0.5f, 0.5f);
        float c   = jg * cg + fg * cacc[i][r];
        cacc[i][r] = c;
        float hs  = og4 * ftanh(c);
#else
        float fg  = fsig2(fmaf(g0[r], NDL, breg[0][i]));
        float jg  = fsig2(fmaf(g1[r], NDL, breg[1][i]));
        float og4 = frcp(0.25f + fexp2(fmaf(g2[r], NDL, breg[2][i])));
        float cg  = fsig2(fmaf(g3[r], NDL, breg[3][i]));
        float c   = jg * cg + fg * cacc[i][r];
        cacc[i][r] = c;
        float s2  = fsig2(c * (-2.0f * LOG2E));
        float hs  = og4 * fmaf(s2, 2.0f, -1.0f);
#endif
        outacc[r] += hs * linreg[i];
        *(unsigned char*)(baseh + (r + hi * 8) * 64) = f2fp8(hs);
      }
      __builtin_amdgcn_sched_barrier(0);
    }

#pragma unroll
    for (int r = 0; r < 8; ++r) {
      float s = outacc[r];
      s += __shfl_xor(s, 1, 32);
      s += __shfl_xor(s, 2, 32);
      s += __shfl_xor(s, 4, 32);
      s += __shfl_xor(s, 8, 32);
      outacc[r] = s;
    }
    if (lane16 == 0) {
#pragma unroll
      for (int r = 0; r < 8; ++r) {
        int row = hi * 8 + r;
        *(float*)(smem + OFF_OUT + (unsigned)(((tile * 16 + row) * 2 + half) * 4)) = outacc[r];
      }
    }
  }

  __syncthreads();
  if (half == 0 && lane < 16) {
    float2 p = *(const float2*)(smem + OFF_OUT + (unsigned)((tile * 16 + lane16) * 8));
    int n = nrow0 + lane16;
    if (n < N_DIM) out[(size_t)(T_DIM - 1) * N_DIM + n] = p.x + p.y + lb;
  }
}

extern "C" void kernel_launch(void* const* d_in, const int* in_sizes, int n_in,
                              void* d_out, int out_size, void* d_ws, size_t ws_size,
                              hipStream_t stream) {
  (void)in_sizes; (void)n_in; (void)out_size; (void)d_ws; (void)ws_size;
  const float* AX     = (const float*)d_in[0];
  const float* W      = (const float*)d_in[1];
  const float* Wf     = (const float*)d_in[2];
  const float* Wj     = (const float*)d_in[3];
  const float* Wc     = (const float*)d_in[4];
  const float* Wo     = (const float*)d_in[5];
  const float* Uf     = (const float*)d_in[6];
  const float* Uj     = (const float*)d_in[7];
  const float* Uc     = (const float*)d_in[8];
  const float* Uo     = (const float*)d_in[9];
  const float* bfv    = (const float*)d_in[10];
  const float* bjv    = (const float*)d_in[11];
  const float* bcv    = (const float*)d_in[12];
  const float* bov    = (const float*)d_in[13];
  const float* h_init = (const float*)d_in[14];
  const float* c_init = (const float*)d_in[15];
  const float* lin_w  = (const float*)d_in[16];
  const float* lin_b  = (const float*)d_in[17];
  float* out = (float*)d_out;

  const int blocks = (N_DIM + 63) / 64;
  wdgcn_fused_fp8_kernel<<<blocks, 256, SMEM_TOTAL, stream>>>(
      AX, W, Wf, Wj, Wc, Wo, Uf, Uj, Uc, Uo,
      bfv, bjv, bcv, bov, h_init, c_init, lin_w, lin_b, out);
}
